// MultiHeadRelAttn_38972533244457
// MI455X (gfx1250) — matrix operand order repaired and checked by reading, not compiled, not run on hardware
//
#include <hip/hip_runtime.h>
#include <hip/hip_bf16.h>

// MultiHeadRelAttn for MI455X (gfx1250, wave32, WMMA bf16).
// B=8, S=1024, DM=1024, H=16, D=64.
//
// Pipeline (all on `stream`, ordered by stream semantics):
//  K1  qkv_proj      : X @ W^T + b  -> split-head bf16 [BH][S][D]   (3 GEMMs via grid.z)
//  K2a qk_scores     : per (b,h): Qh @ Kh^T -> scores (d_out p_attn region, fp32)
//  K2b relk_scores   : per i: Q_i[128x64] @ rel_k[i]^T[64xS] += scores (rel_k read ONCE)
//  K3  softmax       : scale 1/8, mask, softmax rows in place -> p_attn
//  K4a pv_ctx        : per (b,h): P @ Vh -> ctx fp32 [B][S][H][D]
//  K4b relv_ctx      : per i: P_i[128xS] @ rel_v[i][Sx64] += ctx   (rel_v read ONCE)
//  K5  out_proj      : ctx @ Wo^T + bo -> out
//
// Workspace: qh/kh/vh bf16 (3 x 16MB) + ctx fp32 (32MB) = 80MB.
// bf16 tiles already resident in ws are staged to LDS with
// GLOBAL_LOAD_ASYNC_TO_LDS_B128 (ASYNCcnt) when the toolchain exposes it.

#define B_   8
#define S_   1024
#define DM_  1024
#define H_   16
#define D_   64
#define BH_  (B_ * H_)
#define MTOT_ (B_ * S_)

typedef __attribute__((ext_vector_type(16))) __bf16 v16bf;
typedef __attribute__((ext_vector_type(8)))  float  v8f;

union BF16Frag { v16bf v; unsigned int u[8]; };

// --- fp32x2 -> packed bf16 dword -------------------------------------------
#if __has_builtin(__builtin_amdgcn_cvt_pk_bf16_f32)
__device__ inline unsigned int pack2_bf16(float lo, float hi) {
  return __builtin_bit_cast(unsigned int,
                            __builtin_amdgcn_cvt_pk_bf16_f32(lo, hi));
}
#else
__device__ inline unsigned int pack2_bf16(float lo, float hi) {
  unsigned int a = __builtin_bit_cast(unsigned int, lo);
  unsigned int b = __builtin_bit_cast(unsigned int, hi);
  a = (a + 0x7FFFu + ((a >> 16) & 1u)) >> 16;   // round-to-nearest-even
  b = (b + 0x7FFFu + ((b >> 16) & 1u)) >> 16;
  return (a & 0xFFFFu) | (b << 16);
}
#endif

__device__ inline unsigned short to_bf16(float x) {
  unsigned int a = __builtin_bit_cast(unsigned int, x);
  a = (a + 0x7FFFu + ((a >> 16) & 1u)) >> 16;
  return (unsigned short)a;
}

// --- async global -> LDS 16B copy (gfx1250 GLOBAL_LOAD_ASYNC_TO_LDS_B128) --
// Builtin prototype (from hipcc diagnostic): params are pointers to
// `int __attribute__((vector_size(16)))` in AS1 (global) / AS3 (LDS).
#if __has_builtin(__builtin_amdgcn_global_load_async_to_lds_b128)
#define HAVE_ASYNC_LDS 1
typedef int v4i_t __attribute__((vector_size(4 * sizeof(int))));
typedef __attribute__((address_space(1))) v4i_t* gbl_v4i_p;
typedef __attribute__((address_space(3))) v4i_t* lds_v4i_p;
__device__ inline void async_copy16(const void* g, void* l) {
  __builtin_amdgcn_global_load_async_to_lds_b128((gbl_v4i_p)g, (lds_v4i_p)l, 0, 0);
}
__device__ inline void async_wait() {
#if __has_builtin(__builtin_amdgcn_s_wait_asynccnt)
  __builtin_amdgcn_s_wait_asynccnt(0);
#else
  asm volatile("s_wait_asynccnt 0" ::: "memory");
#endif
}
#else
#define HAVE_ASYNC_LDS 0
__device__ inline void async_copy16(const void* g, void* l) {
  *(uint4*)l = *(const uint4*)g;
}
__device__ inline void async_wait() {}
#endif

// --- WMMA fragment loaders (contiguous k order, same for A and B) ----------------------------
// A fragment: 16x32 bf16 tile in LDS, row-major, `ld` elements per row.
// lanes 0-15: row M=lane, K=0..15; lanes 16-31: K=16..31.
__device__ inline v16bf load_a_frag(const unsigned short* lds, int ld) {
  const int lane = threadIdx.x & 31;
  const int m  = lane & 15;
  const int hf = (lane >> 4) << 4;              // 0 or 16
  const unsigned int* row = (const unsigned int*)(lds + m * ld);
  BF16Frag f;
#pragma unroll
  for (int p = 0; p < 8; ++p) f.u[p] = row[(hf >> 1) + p];              // K=hf+2p
  return f.v;
}

// B fragment from LDS stored [n][k] (16 rows of n, `ld` stride): B[k][n]=lds[n][k].
__device__ inline v16bf load_b_frag_nk(const unsigned short* lds, int ld) {
  const int lane = threadIdx.x & 31;
  const int n  = lane & 15;
  const int kb = (lane >> 4) << 4;              // 0 or 16
  const unsigned int* row = (const unsigned int*)(lds + n * ld);
  BF16Frag f;
#pragma unroll
  for (int p = 0; p < 8; ++p) f.u[p] = row[(kb >> 1) + p];
  return f.v;
}

// B fragment from LDS stored [k][n] (32 rows of k, `ld` stride).
__device__ inline v16bf load_b_frag_kn(const unsigned short* lds, int ld) {
  const int lane = threadIdx.x & 31;
  const int n  = lane & 15;
  const int kb = (lane >> 4) << 4;
  BF16Frag f;
#pragma unroll
  for (int p = 0; p < 8; ++p) {
    unsigned int lo = lds[(kb + 2 * p)     * ld + n];
    unsigned int hi = lds[(kb + 2 * p + 1) * ld + n];
    f.u[p] = lo | (hi << 16);
  }
  return f.v;
}

__device__ inline v8f wmma_bf16(v16bf a, v16bf b, v8f c) {
  return __builtin_amdgcn_wmma_f32_16x16x32_bf16(false, a, false, b, (short)0, c,
                                                 false, false);
}

// Stage a ROWSxCOLS fp32 tile (row stride `srcld`) into LDS as bf16, using
// float4 loads (global_load_b128) and packed bf16 stores.
// ITER * 256 * 4 == ROWS * COLS must hold. CCHUNK = COLS/4.
template <int ITER, int CCHUNK>
__device__ inline void stage_f32(unsigned int* dst, const float* src, size_t srcld) {
  const int tid = threadIdx.x;
#pragma unroll
  for (int t = 0; t < ITER; ++t) {
    int idx = tid + t * 256;
    int r = idx / CCHUNK, c4 = (idx % CCHUNK) * 4;
    const float4 f = *(const float4*)(src + (size_t)r * srcld + c4);
    dst[r * (CCHUNK * 2) + (c4 >> 1)]     = pack2_bf16(f.x, f.y);
    dst[r * (CCHUNK * 2) + (c4 >> 1) + 1] = pack2_bf16(f.z, f.w);
  }
}

// ---------------------------------------------------------------------------
// K1: fused QKV projection. grid = (DM/64, MTOT/128, 3)
// ---------------------------------------------------------------------------
__global__ __launch_bounds__(256)
void qkv_proj_kernel(const float* __restrict__ q, const float* __restrict__ k,
                     const float* __restrict__ v,
                     const float* __restrict__ Wq, const float* __restrict__ Wk,
                     const float* __restrict__ Wv,
                     const float* __restrict__ bq, const float* __restrict__ bk,
                     const float* __restrict__ bv,
                     unsigned short* __restrict__ qh, unsigned short* __restrict__ kh,
                     unsigned short* __restrict__ vh) {
  __shared__ unsigned short As[128 * 32];
  __shared__ unsigned short Bs[64 * 32];
  const float* X; const float* W; const float* bias; unsigned short* out;
  if (blockIdx.z == 0)      { X = q; W = Wq; bias = bq; out = qh; }
  else if (blockIdx.z == 1) { X = k; W = Wk; bias = bk; out = kh; }
  else                      { X = v; W = Wv; bias = bv; out = vh; }

  const int tid = threadIdx.x, lane = tid & 31, w = tid >> 5;
  const int wm = (w & 3) * 32, wn = (w >> 2) * 32;
  const int gm0 = blockIdx.y * 128, gn0 = blockIdx.x * 64;

  v8f acc[2][2]; v8f z = {};
#pragma unroll
  for (int a = 0; a < 2; ++a)
#pragma unroll
    for (int b = 0; b < 2; ++b) acc[a][b] = z;

  for (int k0 = 0; k0 < DM_; k0 += 32) {
    stage_f32<4, 8>((unsigned int*)As, X + (size_t)gm0 * DM_ + k0, DM_);
    stage_f32<2, 8>((unsigned int*)Bs, W + (size_t)gn0 * DM_ + k0, DM_);
    if (k0 + 32 < DM_) {                          // hint next tile -> global_prefetch
      __builtin_prefetch(X + (size_t)(gm0 + (tid >> 1)) * DM_ + k0 + 32, 0, 0);
      __builtin_prefetch(W + (size_t)(gn0 + (tid & 63)) * DM_ + k0 + 32, 0, 0);
    }
    __syncthreads();
    v16bf a0 = load_a_frag(As + (wm + 0)  * 32, 32);
    v16bf a1 = load_a_frag(As + (wm + 16) * 32, 32);
    v16bf b0 = load_b_frag_nk(Bs + (wn + 0)  * 32, 32);
    v16bf b1 = load_b_frag_nk(Bs + (wn + 16) * 32, 32);
    acc[0][0] = wmma_bf16(a0, b0, acc[0][0]);
    acc[0][1] = wmma_bf16(a0, b1, acc[0][1]);
    acc[1][0] = wmma_bf16(a1, b0, acc[1][0]);
    acc[1][1] = wmma_bf16(a1, b1, acc[1][1]);
    __syncthreads();
  }

  const int colbase = lane & 15, rowoff = (lane >> 4) << 3;
#pragma unroll
  for (int ti = 0; ti < 2; ++ti)
#pragma unroll
    for (int tj = 0; tj < 2; ++tj)
#pragma unroll
      for (int r = 0; r < 8; ++r) {
        int m = gm0 + wm + ti * 16 + rowoff + r;
        int n = gn0 + wn + tj * 16 + colbase;
        float val = acc[ti][tj][r] + bias[n];
        int bb = m >> 10, s = m & 1023;           // m = b*S + s
        int hh = n >> 6,  d = n & 63;             // n = h*D + d
        out[((size_t)(bb * H_ + hh) * S_ + s) * D_ + d] = to_bf16(val);
      }
}

// ---------------------------------------------------------------------------
// K2a: scores = Qh @ Kh^T per (b,h). grid = (S/64, S/128, BH)
// Qh/Kh tiles are bf16 in ws -> staged with async-to-LDS b128.
// ---------------------------------------------------------------------------
__global__ __launch_bounds__(256)
void qk_scores_kernel(const unsigned short* __restrict__ qh,
                      const unsigned short* __restrict__ kh,
                      float* __restrict__ scores) {
  __shared__ unsigned short As[128 * 32];
  __shared__ unsigned short Bs[64 * 32];
  const int tid = threadIdx.x, lane = tid & 31, w = tid >> 5;
  const int wm = (w & 3) * 32, wn = (w >> 2) * 32;
  const int bh = blockIdx.z;
  const int gm0 = blockIdx.y * 128, gn0 = blockIdx.x * 64;
  const unsigned short* qb = qh + (size_t)bh * S_ * D_;
  const unsigned short* kb = kh + (size_t)bh * S_ * D_;

  v8f acc[2][2]; v8f z = {};
#pragma unroll
  for (int a = 0; a < 2; ++a)
#pragma unroll
    for (int b = 0; b < 2; ++b) acc[a][b] = z;

  for (int k0 = 0; k0 < D_; k0 += 32) {
#pragma unroll
    for (int t = 0; t < 2; ++t) {                 // A: 128x32 bf16, 16B chunks
      int idx = tid + t * 256;
      int r = idx >> 2, c8 = (idx & 3) << 3;
      async_copy16(qb + (size_t)(gm0 + r) * D_ + k0 + c8, As + r * 32 + c8);
    }
    {                                             // B: 64x32 bf16
      int r = tid >> 2, c8 = (tid & 3) << 3;
      async_copy16(kb + (size_t)(gn0 + r) * D_ + k0 + c8, Bs + r * 32 + c8);
    }
    async_wait();
    __syncthreads();
    v16bf a0 = load_a_frag(As + (wm + 0)  * 32, 32);
    v16bf a1 = load_a_frag(As + (wm + 16) * 32, 32);
    v16bf b0 = load_b_frag_nk(Bs + (wn + 0)  * 32, 32);
    v16bf b1 = load_b_frag_nk(Bs + (wn + 16) * 32, 32);
    acc[0][0] = wmma_bf16(a0, b0, acc[0][0]);
    acc[0][1] = wmma_bf16(a0, b1, acc[0][1]);
    acc[1][0] = wmma_bf16(a1, b0, acc[1][0]);
    acc[1][1] = wmma_bf16(a1, b1, acc[1][1]);
    __syncthreads();
  }

  float* sc = scores + (size_t)bh * S_ * S_;
  const int colbase = lane & 15, rowoff = (lane >> 4) << 3;
#pragma unroll
  for (int ti = 0; ti < 2; ++ti)
#pragma unroll
    for (int tj = 0; tj < 2; ++tj)
#pragma unroll
      for (int r = 0; r < 8; ++r) {
        int m = gm0 + wm + ti * 16 + rowoff + r;
        int n = gn0 + wn + tj * 16 + colbase;
        sc[(size_t)m * S_ + n] = acc[ti][tj][r];
      }
}

// ---------------------------------------------------------------------------
// K2b: scores[:, i, :] += Q_i[128x64] @ rel_k[i]^T. grid = (S/64, 1, S)
// rel_k (256MB) streamed exactly once.
// ---------------------------------------------------------------------------
__global__ __launch_bounds__(256)
void relk_scores_kernel(const unsigned short* __restrict__ qh,
                        const float* __restrict__ rel_k,
                        float* __restrict__ scores) {
  __shared__ unsigned short As[128 * 32];
  __shared__ unsigned short Bs[64 * 32];
  const int tid = threadIdx.x, lane = tid & 31, w = tid >> 5;
  const int wm = (w & 3) * 32, wn = (w >> 2) * 32;
  const int i = blockIdx.z;
  const int gn0 = blockIdx.x * 64;

  v8f acc[2][2]; v8f z = {};
#pragma unroll
  for (int a = 0; a < 2; ++a)
#pragma unroll
    for (int b = 0; b < 2; ++b) acc[a][b] = z;

  for (int k0 = 0; k0 < D_; k0 += 32) {
#pragma unroll
    for (int t = 0; t < 2; ++t) {                 // A: rows = bh, strided gather
      int idx = tid + t * 256;
      int r = idx >> 2, c8 = (idx & 3) << 3;
      async_copy16(qh + ((size_t)r * S_ + i) * D_ + k0 + c8, As + r * 32 + c8);
    }
    stage_f32<2, 8>((unsigned int*)Bs,            // B: rel_k[i][j][d] fp32->bf16
                    rel_k + ((size_t)i * S_ + gn0) * D_ + k0, D_);
    async_wait();
    __syncthreads();
    v16bf a0 = load_a_frag(As + (wm + 0)  * 32, 32);
    v16bf a1 = load_a_frag(As + (wm + 16) * 32, 32);
    v16bf b0 = load_b_frag_nk(Bs + (wn + 0)  * 32, 32);
    v16bf b1 = load_b_frag_nk(Bs + (wn + 16) * 32, 32);
    acc[0][0] = wmma_bf16(a0, b0, acc[0][0]);
    acc[0][1] = wmma_bf16(a0, b1, acc[0][1]);
    acc[1][0] = wmma_bf16(a1, b0, acc[1][0]);
    acc[1][1] = wmma_bf16(a1, b1, acc[1][1]);
    __syncthreads();
  }

  const int colbase = lane & 15, rowoff = (lane >> 4) << 3;
#pragma unroll
  for (int ti = 0; ti < 2; ++ti)
#pragma unroll
    for (int tj = 0; tj < 2; ++tj)
#pragma unroll
      for (int r = 0; r < 8; ++r) {
        int m = wm + ti * 16 + rowoff + r;        // bh
        int n = gn0 + wn + tj * 16 + colbase;     // j
        float* dst = scores + ((size_t)m * S_ + i) * S_ + n;
        *dst += acc[ti][tj][r];
      }
}

// ---------------------------------------------------------------------------
// K3: softmax over j with scale + mask, in place. grid = (S, BH), 256 thr.
// ---------------------------------------------------------------------------
__global__ __launch_bounds__(256)
void softmax_kernel(float* __restrict__ p, const unsigned char* __restrict__ mask) {
  const int i = blockIdx.x, bh = blockIdx.y;
  const int b = bh >> 4;                          // bh / H
  float* row = p + ((size_t)bh * S_ + i) * S_;
  const unsigned char* mrow = mask + ((size_t)b * S_ + i) * S_;
  const int tid = threadIdx.x;

  float4 x = ((const float4*)row)[tid];
  uchar4 mk = ((const uchar4*)mrow)[tid];
  const float sc = 0.125f;                        // 1/sqrt(D)
  const float NEG = -__builtin_inff();
  float v0 = mk.x ? NEG : x.x * sc;
  float v1 = mk.y ? NEG : x.y * sc;
  float v2 = mk.z ? NEG : x.z * sc;
  float v3 = mk.w ? NEG : x.w * sc;

  float mx = fmaxf(fmaxf(v0, v1), fmaxf(v2, v3));
#pragma unroll
  for (int off = 16; off > 0; off >>= 1) mx = fmaxf(mx, __shfl_xor(mx, off));
  __shared__ float redm[8];
  if ((tid & 31) == 0) redm[tid >> 5] = mx;
  __syncthreads();
  mx = redm[0];
#pragma unroll
  for (int qq = 1; qq < 8; ++qq) mx = fmaxf(mx, redm[qq]);

  float e0 = __expf(v0 - mx), e1 = __expf(v1 - mx);
  float e2 = __expf(v2 - mx), e3 = __expf(v3 - mx);
  float sum = e0 + e1 + e2 + e3;
#pragma unroll
  for (int off = 16; off > 0; off >>= 1) sum += __shfl_xor(sum, off);
  __shared__ float reds[8];
  if ((tid & 31) == 0) reds[tid >> 5] = sum;
  __syncthreads();
  sum = reds[0] + reds[1] + reds[2] + reds[3] + reds[4] + reds[5] + reds[6] + reds[7];
  float inv = 1.0f / sum;
  ((float4*)row)[tid] = make_float4(e0 * inv, e1 * inv, e2 * inv, e3 * inv);
}

// ---------------------------------------------------------------------------
// K4a: ctx = P @ Vh per (b,h). grid = (1, S/128, BH). ctx layout [B][S][H][D].
// ---------------------------------------------------------------------------
__global__ __launch_bounds__(256)
void pv_ctx_kernel(const float* __restrict__ p, const unsigned short* __restrict__ vh,
                   float* __restrict__ ctx) {
  __shared__ unsigned short As[128 * 32];
  __shared__ unsigned short Bs[32 * 64];
  const int tid = threadIdx.x, lane = tid & 31, w = tid >> 5;
  const int wm = (w & 3) * 32, wn = (w >> 2) * 32;
  const int bh = blockIdx.z;
  const int gm0 = blockIdx.y * 128;
  const float* pb = p + (size_t)bh * S_ * S_;
  const unsigned short* vb = vh + (size_t)bh * S_ * D_;

  v8f acc[2][2]; v8f z = {};
#pragma unroll
  for (int a = 0; a < 2; ++a)
#pragma unroll
    for (int b = 0; b < 2; ++b) acc[a][b] = z;

  for (int k0 = 0; k0 < S_; k0 += 32) {
    stage_f32<4, 8>((unsigned int*)As, pb + (size_t)gm0 * S_ + k0, S_);
    {                                             // B: Vh 32x64 bf16, async
      int r = tid >> 3, c8 = (tid & 7) << 3;
      async_copy16(vb + (size_t)(k0 + r) * D_ + c8, Bs + r * 64 + c8);
    }
    if (k0 + 32 < S_)
      __builtin_prefetch(pb + (size_t)(gm0 + (tid >> 1)) * S_ + k0 + 32, 0, 0);
    async_wait();
    __syncthreads();
    v16bf a0 = load_a_frag(As + (wm + 0)  * 32, 32);
    v16bf a1 = load_a_frag(As + (wm + 16) * 32, 32);
    v16bf b0 = load_b_frag_kn(Bs + (wn + 0),  64);
    v16bf b1 = load_b_frag_kn(Bs + (wn + 16), 64);
    acc[0][0] = wmma_bf16(a0, b0, acc[0][0]);
    acc[0][1] = wmma_bf16(a0, b1, acc[0][1]);
    acc[1][0] = wmma_bf16(a1, b0, acc[1][0]);
    acc[1][1] = wmma_bf16(a1, b1, acc[1][1]);
    __syncthreads();
  }

  const int bb = bh >> 4, hh = bh & 15;
  const int colbase = lane & 15, rowoff = (lane >> 4) << 3;
#pragma unroll
  for (int ti = 0; ti < 2; ++ti)
#pragma unroll
    for (int tj = 0; tj < 2; ++tj)
#pragma unroll
      for (int r = 0; r < 8; ++r) {
        int ii = gm0 + wm + ti * 16 + rowoff + r;
        int d  = wn + tj * 16 + colbase;
        ctx[(((size_t)bb * S_ + ii) * H_ + hh) * D_ + d] = acc[ti][tj][r];
      }
}

// ---------------------------------------------------------------------------
// K4b: ctx[:, i, :] += P_i[128xS] @ rel_v[i][Sx64]. grid = (1, 1, S).
// rel_v (256MB) streamed exactly once.
// ---------------------------------------------------------------------------
__global__ __launch_bounds__(256)
void relv_ctx_kernel(const float* __restrict__ p, const float* __restrict__ rel_v,
                     float* __restrict__ ctx) {
  __shared__ unsigned short As[128 * 32];
  __shared__ unsigned short Bs[32 * 64];
  const int tid = threadIdx.x, lane = tid & 31, w = tid >> 5;
  const int wm = (w & 3) * 32, wn = (w >> 2) * 32;
  const int i = blockIdx.z;

  v8f acc[2][2]; v8f z = {};
#pragma unroll
  for (int a = 0; a < 2; ++a)
#pragma unroll
    for (int b = 0; b < 2; ++b) acc[a][b] = z;

  for (int k0 = 0; k0 < S_; k0 += 32) {
#pragma unroll
    for (int t = 0; t < 4; ++t) {                 // A: P rows = bh (stride S*S)
      int idx = tid + t * 256;
      int r = idx >> 3, c4 = (idx & 7) << 2;
      const float4 f = *(const float4*)(p + ((size_t)r * S_ + i) * S_ + k0 + c4);
      ((unsigned int*)As)[r * 16 + (c4 >> 1)]     = pack2_bf16(f.x, f.y);
      ((unsigned int*)As)[r * 16 + (c4 >> 1) + 1] = pack2_bf16(f.z, f.w);
    }
    stage_f32<2, 16>((unsigned int*)Bs,           // B: rel_v[i][j][d] [k][n]
                     rel_v + ((size_t)i * S_ + k0) * D_, D_);
    if (k0 + 32 < S_)
      __builtin_prefetch(rel_v + ((size_t)i * S_ + k0 + 32 + (tid >> 3)) * D_, 0, 0);
    __syncthreads();
    v16bf a0 = load_a_frag(As + (wm + 0)  * 32, 32);
    v16bf a1 = load_a_frag(As + (wm + 16) * 32, 32);
    v16bf b0 = load_b_frag_kn(Bs + (wn + 0),  64);
    v16bf b1 = load_b_frag_kn(Bs + (wn + 16), 64);
    acc[0][0] = wmma_bf16(a0, b0, acc[0][0]);
    acc[0][1] = wmma_bf16(a0, b1, acc[0][1]);
    acc[1][0] = wmma_bf16(a1, b0, acc[1][0]);
    acc[1][1] = wmma_bf16(a1, b1, acc[1][1]);
    __syncthreads();
  }

  const int colbase = lane & 15, rowoff = (lane >> 4) << 3;
#pragma unroll
  for (int ti = 0; ti < 2; ++ti)
#pragma unroll
    for (int tj = 0; tj < 2; ++tj)
#pragma unroll
      for (int r = 0; r < 8; ++r) {
        int m = wm + ti * 16 + rowoff + r;        // bh
        int d = wn + tj * 16 + colbase;
        int bb = m >> 4, hh = m & 15;
        float* dst = &ctx[(((size_t)bb * S_ + i) * H_ + hh) * D_ + d];
        *dst += acc[ti][tj][r];
      }
}

// ---------------------------------------------------------------------------
// K5: out = ctx @ Wo^T + bo. grid = (DM/64, MTOT/128)
// ---------------------------------------------------------------------------
__global__ __launch_bounds__(256)
void out_proj_kernel(const float* __restrict__ ctx, const float* __restrict__ Wo,
                     const float* __restrict__ bo, float* __restrict__ out) {
  __shared__ unsigned short As[128 * 32];
  __shared__ unsigned short Bs[64 * 32];
  const int tid = threadIdx.x, lane = tid & 31, w = tid >> 5;
  const int wm = (w & 3) * 32, wn = (w >> 2) * 32;
  const int gm0 = blockIdx.y * 128, gn0 = blockIdx.x * 64;

  v8f acc[2][2]; v8f z = {};
#pragma unroll
  for (int a = 0; a < 2; ++a)
#pragma unroll
    for (int b = 0; b < 2; ++b) acc[a][b] = z;

  for (int k0 = 0; k0 < DM_; k0 += 32) {
    stage_f32<4, 8>((unsigned int*)As, ctx + (size_t)gm0 * DM_ + k0, DM_);
    stage_f32<2, 8>((unsigned int*)Bs, Wo + (size_t)gn0 * DM_ + k0, DM_);
    if (k0 + 32 < DM_)
      __builtin_prefetch(ctx + (size_t)(gm0 + (tid >> 1)) * DM_ + k0 + 32, 0, 0);
    __syncthreads();
    v16bf a0 = load_a_frag(As + (wm + 0)  * 32, 32);
    v16bf a1 = load_a_frag(As + (wm + 16) * 32, 32);
    v16bf b0 = load_b_frag_nk(Bs + (wn + 0)  * 32, 32);
    v16bf b1 = load_b_frag_nk(Bs + (wn + 16) * 32, 32);
    acc[0][0] = wmma_bf16(a0, b0, acc[0][0]);
    acc[0][1] = wmma_bf16(a0, b1, acc[0][1]);
    acc[1][0] = wmma_bf16(a1, b0, acc[1][0]);
    acc[1][1] = wmma_bf16(a1, b1, acc[1][1]);
    __syncthreads();
  }

  const int colbase = lane & 15, rowoff = (lane >> 4) << 3;
#pragma unroll
  for (int ti = 0; ti < 2; ++ti)
#pragma unroll
    for (int tj = 0; tj < 2; ++tj)
#pragma unroll
      for (int r = 0; r < 8; ++r) {
        int m = gm0 + wm + ti * 16 + rowoff + r;
        int n = gn0 + wn + tj * 16 + colbase;
        out[(size_t)m * DM_ + n] = acc[ti][tj][r] + bo[n];
      }
}

// ---------------------------------------------------------------------------
extern "C" void kernel_launch(void* const* d_in, const int* in_sizes, int n_in,
                              void* d_out, int out_size, void* d_ws, size_t ws_size,
                              hipStream_t stream) {
  const float* q  = (const float*)d_in[0];
  const float* k  = (const float*)d_in[1];
  const float* v  = (const float*)d_in[2];
  const unsigned char* mask = (const unsigned char*)d_in[3];
  const float* rel_k = (const float*)d_in[4];
  const float* rel_v = (const float*)d_in[5];
  const float* Wq = (const float*)d_in[6];
  const float* bq = (const float*)d_in[7];
  const float* Wk = (const float*)d_in[8];
  const float* bk = (const float*)d_in[9];
  const float* Wv = (const float*)d_in[10];
  const float* bv = (const float*)d_in[11];
  const float* Wo = (const float*)d_in[12];
  const float* bo = (const float*)d_in[13];

  float* out    = (float*)d_out;
  float* p_attn = out + (size_t)B_ * S_ * DM_;    // scores buffer, then p_attn

  unsigned short* qh = (unsigned short*)d_ws;     // bf16 [BH][S][D]
  unsigned short* kh = qh + (size_t)BH_ * S_ * D_;
  unsigned short* vh = kh + (size_t)BH_ * S_ * D_;
  float* ctx = (float*)(vh + (size_t)BH_ * S_ * D_);  // fp32 [B][S][H][D]

  dim3 blk(256, 1, 1);
  qkv_proj_kernel<<<dim3(DM_ / 64, MTOT_ / 128, 3), blk, 0, stream>>>(
      q, k, v, Wq, Wk, Wv, bq, bk, bv, qh, kh, vh);
  qk_scores_kernel<<<dim3(S_ / 64, S_ / 128, BH_), blk, 0, stream>>>(qh, kh, p_attn);
  relk_scores_kernel<<<dim3(S_ / 64, 1, S_), blk, 0, stream>>>(qh, rel_k, p_attn);
  softmax_kernel<<<dim3(S_, BH_, 1), blk, 0, stream>>>(p_attn, mask);
  pv_ctx_kernel<<<dim3(1, S_ / 128, BH_), blk, 0, stream>>>(p_attn, vh, ctx);
  relv_ctx_kernel<<<dim3(1, 1, S_), blk, 0, stream>>>(p_attn, rel_v, ctx);
  out_proj_kernel<<<dim3(DM_ / 64, MTOT_ / 128, 1), blk, 0, stream>>>(ctx, Wo, bo, out);
}
